// TransformerBlockMC_87600152969888
// MI455X (gfx1250) — matrix operand order repaired and checked by reading, not compiled, not run on hardware
//
#include <hip/hip_runtime.h>
#include <hip/hip_bf16.h>

// ---------------------------------------------------------------------------
// Types for CDNA5 WMMA (wave32, 16x16x32 bf16 -> f32)
// ---------------------------------------------------------------------------
typedef __bf16 bf16_t;
typedef __attribute__((ext_vector_type(16))) __bf16 v16bf;
typedef __attribute__((ext_vector_type(8)))  __bf16 v8bf;
typedef __attribute__((ext_vector_type(4)))  __bf16 v4bf;
typedef __attribute__((ext_vector_type(8)))  float  v8f;

union V16 { v16bf v; v8bf h[2]; };

__device__ __forceinline__ bf16_t to_bf16(float f) {
    union { float f; unsigned u; } c; c.f = f;
    unsigned r = (c.u + 0x7FFFu + ((c.u >> 16) & 1u)) >> 16;
    union { unsigned short s; bf16_t b; } o; o.s = (unsigned short)r;
    return o.b;
}
__device__ __forceinline__ float from_bf16(bf16_t b) {
    union { unsigned short s; bf16_t b; } i; i.b = b;
    union { unsigned u; float f; } o; o.u = ((unsigned)i.s) << 16;
    return o.f;
}

__device__ __forceinline__ v8f wmma_bf16(const V16& a, const V16& b, v8f c) {
    return __builtin_amdgcn_wmma_f32_16x16x32_bf16(
        /*neg_a=*/false, a.v, /*neg_b=*/false, b.v,
        /*c_mod=*/(short)0, c, /*reuse_a=*/false, /*reuse_b=*/false);
}

// ---------------------------------------------------------------------------
// CDNA5 async global->LDS copies (ASYNCcnt-tracked), guarded by __has_builtin.
// The builtin's params are int4 pointers in AS(1) / AS(3) (per hipcc
// diagnostic: "__vector_size__(4*sizeof(int)) int __device__ *").
// ---------------------------------------------------------------------------
#if defined(__has_builtin)
#if __has_builtin(__builtin_amdgcn_global_load_async_to_lds_b128)
#define HAVE_ASYNC_COPY 1
#endif
#endif

#ifdef HAVE_ASYNC_COPY
typedef int v4i_ __attribute__((vector_size(4 * sizeof(int))));
#define GAS1(p) ((__attribute__((address_space(1))) v4i_*)(unsigned long long)(p))
#define LAS3(p) ((__attribute__((address_space(3))) v4i_*)(unsigned int)(unsigned long long)(p))
__device__ __forceinline__ void async_copy_b128(const bf16_t* g, bf16_t* l) {
    __builtin_amdgcn_global_load_async_to_lds_b128(GAS1(g), LAS3(l), 0, 0);
}
#if __has_builtin(__builtin_amdgcn_s_wait_asynccnt)
#define WAIT_ASYNC(n) __builtin_amdgcn_s_wait_asynccnt(n)
#else
#define WAIT_ASYNC(n) asm volatile("s_wait_asynccnt %0" ::"i"(n))
#endif
#endif

// ---------------------------------------------------------------------------
// Problem constants
// ---------------------------------------------------------------------------
#define B_  2
#define S_  2048
#define D_  2048
#define H_  16
#define HD_ 128
#define F_  8192
#define M_  (B_ * S_)          // 4096 token rows
#define EPS_ 1e-6f
#define RES_SCALE_ 0.5773502691896258f   // 1/sqrt(3), layer_idx = 1
#define ATTN_SCALE_ 0.08838834764831845f // 1/sqrt(128)

// ---------------------------------------------------------------------------
// fp32 -> bf16 cast (weights), 4 elems/thread, vectorized
// ---------------------------------------------------------------------------
__global__ __launch_bounds__(256) void k_cast_bf16(const float* __restrict__ in,
                                                   bf16_t* __restrict__ out) {
    size_t i = ((size_t)blockIdx.x * 256 + threadIdx.x) * 4;
    float4 v = *(const float4*)(in + i);
    v4bf o;
    o[0] = to_bf16(v.x); o[1] = to_bf16(v.y);
    o[2] = to_bf16(v.z); o[3] = to_bf16(v.w);
    *(v4bf*)(out + i) = o;
}

// ---------------------------------------------------------------------------
// RMSNorm: one block per token row (D=2048), fp32 in -> bf16 out
// ---------------------------------------------------------------------------
__global__ __launch_bounds__(256) void k_rmsnorm_bf16(const float* __restrict__ x,
                                                      const float* __restrict__ w,
                                                      bf16_t* __restrict__ out) {
    __shared__ float red[8];
    int row = blockIdx.x;
    const float* xr = x + (size_t)row * D_;
    float s = 0.f;
    for (int c = threadIdx.x; c < D_; c += 256) { float v = xr[c]; s += v * v; }
    for (int off = 16; off > 0; off >>= 1) s += __shfl_xor(s, off, 32);
    if ((threadIdx.x & 31) == 0) red[threadIdx.x >> 5] = s;
    __syncthreads();
    if (threadIdx.x == 0) {
        float t = 0.f;
        for (int i = 0; i < 8; i++) t += red[i];
        red[0] = t;
    }
    __syncthreads();
    float inv = rsqrtf(red[0] * (1.0f / D_) + EPS_);
    bf16_t* orow = out + (size_t)row * D_;
    for (int c = threadIdx.x; c < D_; c += 256)
        orow[c] = to_bf16(xr[c] * inv * w[c]);
}

// ---------------------------------------------------------------------------
// Generic bf16 WMMA GEMM:  C[M,N] = A[M,K] * W[N,K]^T  (both K-contiguous)
// Block tile 128x128, BK=32, 8 waves (2x4), each wave 64x32 = 4x2 WMMA accums.
// Async double-buffered LDS pipeline when the toolchain exposes
// global_load_async_to_lds_b128; synchronous single-buffer path otherwise.
// Epilogues: 1 = bf16 store, 2 = fp32 residual (out = res + RES_SCALE*acc),
//            3 = bf16 silu(acc), 4 = bf16 acc * aux_bf16[idx]
// ---------------------------------------------------------------------------
#define BM 128
#define BN 128
#define BK 32
#define LP 40   // LDS pitch in halves (32 + 8 pad; 80B = 16B-aligned rows)

__device__ __forceinline__ void gemm_step(const bf16_t* __restrict__ As,
                                          const bf16_t* __restrict__ Bs,
                                          int m0, int n0, int l16, int lhalf,
                                          v8f (&acc)[4][2]) {
    // B fragments: lane%16 -> N col, K chunks at (lane/16)*8 and +16
    V16 bf[2];
    #pragma unroll
    for (int j = 0; j < 2; j++) {
        int n = n0 + j * 16 + l16;
        int k0 = lhalf * 8;
        bf[j].h[0] = *(const v8bf*)&Bs[n * LP + k0];
        bf[j].h[1] = *(const v8bf*)&Bs[n * LP + k0 + 16];
    }
    // A fragments: lane%16 -> M row, K chunks at (lane/16)*8 and +16
    #pragma unroll
    for (int i = 0; i < 4; i++) {
        V16 af;
        int m = m0 + i * 16 + l16;
        int k0 = lhalf * 8;
        af.h[0] = *(const v8bf*)&As[m * LP + k0];
        af.h[1] = *(const v8bf*)&As[m * LP + k0 + 16];
        #pragma unroll
        for (int j = 0; j < 2; j++)
            acc[i][j] = wmma_bf16(af, bf[j], acc[i][j]);
    }
}

template <int EPI>
__global__ __launch_bounds__(256)
void k_gemm_bf16(const bf16_t* __restrict__ A, const bf16_t* __restrict__ W,
                 void* __restrict__ Cout, const float* __restrict__ res,
                 const bf16_t* __restrict__ aux, int Mdim, int Ndim, int Kdim) {
    __shared__ bf16_t As[2][BM * LP];
    __shared__ bf16_t Bs[2][BN * LP];

    const int tid = threadIdx.x;
    const int wave = tid >> 5, lane = tid & 31;
    const int l16 = lane & 15, lhalf = lane >> 4;
    const int wm = wave >> 2, wn = wave & 3;        // 2 x 4 wave grid
    const int bm0 = blockIdx.y * BM, bn0 = blockIdx.x * BN;
    const int m0 = wm * 64, n0 = wn * 32;

    const int arow = tid >> 1;                      // 0..127
    const int acol = (tid & 1) * 16;                // 0 or 16 (halves)

    v8f acc[4][2] = {};

#ifdef HAVE_ASYNC_COPY
    auto issue_tile = [&](int kt, int buf) {
        const bf16_t* ga = A + (size_t)(bm0 + arow) * Kdim + kt + acol;
        const bf16_t* gb = W + (size_t)(bn0 + arow) * Kdim + kt + acol;
        bf16_t* la = &As[buf][arow * LP + acol];
        bf16_t* lb = &Bs[buf][arow * LP + acol];
        async_copy_b128(ga, la);
        async_copy_b128(ga + 8, la + 8);
        async_copy_b128(gb, lb);
        async_copy_b128(gb + 8, lb + 8);
    };
    const int nt = Kdim / BK;
    issue_tile(0, 0);
    for (int t = 0; t < nt - 1; t++) {
        const int cur = t & 1;
        issue_tile((t + 1) * BK, cur ^ 1);  // prefetch next tile into other buf
        WAIT_ASYNC(4);                      // wait for tile t (4 pending = next)
        __syncthreads();
        gemm_step(As[cur], Bs[cur], m0, n0, l16, lhalf, acc);
        __syncthreads();                    // all done reading buf cur before
                                            // iteration t+1 overwrites it
    }
    WAIT_ASYNC(0);
    __syncthreads();
    gemm_step(As[(nt - 1) & 1], Bs[(nt - 1) & 1], m0, n0, l16, lhalf, acc);
#else
    for (int kt = 0; kt < Kdim; kt += BK) {
        __syncthreads();
        const bf16_t* ga = A + (size_t)(bm0 + arow) * Kdim + kt + acol;
        const bf16_t* gb = W + (size_t)(bn0 + arow) * Kdim + kt + acol;
        *(v8bf*)&As[0][arow * LP + acol]     = *(const v8bf*)(ga);
        *(v8bf*)&As[0][arow * LP + acol + 8] = *(const v8bf*)(ga + 8);
        *(v8bf*)&Bs[0][arow * LP + acol]     = *(const v8bf*)(gb);
        *(v8bf*)&Bs[0][arow * LP + acol + 8] = *(const v8bf*)(gb + 8);
        if (kt + BK < Kdim) {
            __builtin_prefetch((const void*)(ga + BK), 0, 1);
            __builtin_prefetch((const void*)(gb + BK), 0, 1);
        }
        __syncthreads();
        gemm_step(As[0], Bs[0], m0, n0, l16, lhalf, acc);
    }
#endif

    // Epilogue. C layout: VGPR r -> row (r + (lane/16)*8), col = lane%16.
    #pragma unroll
    for (int i = 0; i < 4; i++) {
        #pragma unroll
        for (int j = 0; j < 2; j++) {
            int n = bn0 + n0 + j * 16 + l16;
            int mb = bm0 + m0 + i * 16 + lhalf * 8;
            #pragma unroll
            for (int r = 0; r < 8; r++) {
                float v = acc[i][j][r];
                size_t idx = (size_t)(mb + r) * Ndim + n;
                if constexpr (EPI == 1) {
                    ((bf16_t*)Cout)[idx] = to_bf16(v);
                } else if constexpr (EPI == 2) {
                    ((float*)Cout)[idx] = res[idx] + RES_SCALE_ * v;
                } else if constexpr (EPI == 3) {
                    float sv = v / (1.0f + __expf(-v));
                    ((bf16_t*)Cout)[idx] = to_bf16(sv);
                } else {
                    float g = from_bf16(aux[idx]);
                    ((bf16_t*)Cout)[idx] = to_bf16(g * v);
                }
            }
        }
    }
}

// ---------------------------------------------------------------------------
// RoPE in-place on bf16 tensor laid out [B, S, H, HD]; one thread per pair.
// ---------------------------------------------------------------------------
__global__ __launch_bounds__(256) void k_rope_bf16(bf16_t* __restrict__ t) {
    int p = blockIdx.x * 256 + threadIdx.x;     // (b,s,h)*64 + hd2
    int hd2 = p & 63;
    int s = (p >> 10) & (S_ - 1);
    size_t base = (size_t)(p >> 6) * HD_;
    float freq = __expf(-(2.0f * hd2 / (float)HD_) * 9.210340371976184f); // ln(1e4)
    float ang = (float)s * freq;
    float sn, cs;
    __sincosf(ang, &sn, &cs);
    float t1 = from_bf16(t[base + 2 * hd2]);
    float t2 = from_bf16(t[base + 2 * hd2 + 1]);
    t[base + 2 * hd2]     = to_bf16(t1 * cs - t2 * sn);
    t[base + 2 * hd2 + 1] = to_bf16(t1 * sn + t2 * cs);
}

// ---------------------------------------------------------------------------
// Flash attention (causal). Q/K/V/O bf16 in [B,S,H,HD] layout.
// Grid: (S/128 q-tiles, B*H). 8 waves x 16 q-rows. Key tiles of 32.
// K staged in LDS row-major (async when available), V staged transposed,
// P restaged through per-wave LDS scratch to get A-fragment layout.
// ---------------------------------------------------------------------------
#define KP 136  // K-tile LDS pitch (272B, multiple of 16B)
#define VP 40   // V^T / P LDS pitch

__global__ __launch_bounds__(256)
void k_attn(const bf16_t* __restrict__ Q, const bf16_t* __restrict__ K,
            const bf16_t* __restrict__ V, bf16_t* __restrict__ O) {
    __shared__ bf16_t Ks[32 * KP];          // [key][hd]
    __shared__ bf16_t Vt[HD_ * VP];         // [hd][key]
    __shared__ bf16_t Ps[8 * 16 * VP];      // per-wave P scratch [m][key]

    const int tid = threadIdx.x, wave = tid >> 5, lane = tid & 31;
    const int l16 = lane & 15, lhalf = lane >> 4;
    const int bh = blockIdx.y, b = bh >> 4, h = bh & 15;
    const int q0 = blockIdx.x * 128 + wave * 16;
    const size_t rs = (size_t)H_ * HD_;     // row stride between s
    const size_t head = ((size_t)b * S_ * H_ + h) * HD_;

    // Q fragments (16 rows x 128 hd = 4 A-frags over the contraction dim)
    V16 qf[4];
    {
        const bf16_t* qr = Q + head + (size_t)(q0 + l16) * rs;
        #pragma unroll
        for (int c = 0; c < 4; c++) {
            int k0 = c * 32 + lhalf * 8;
            qf[c].h[0] = *(const v8bf*)(qr + k0);
            qf[c].h[1] = *(const v8bf*)(qr + k0 + 16);
        }
    }

    v8f oacc[8] = {};
    float mrow[8], lrow[8];
    #pragma unroll
    for (int r = 0; r < 8; r++) { mrow[r] = -3.0e38f; lrow[r] = 0.f; }

    const int nkt = (blockIdx.x + 1) * 4;   // 32-key tiles up to causal frontier
    for (int kt = 0; kt < nkt; kt++) {
        const int key0 = kt * 32;
        __syncthreads();
        {   // cooperative stage of K (as-is) and V (transposed)
            int krow = tid >> 3, kcol = (tid & 7) * 16;
            const bf16_t* gk = K + head + (size_t)(key0 + krow) * rs + kcol;
#ifdef HAVE_ASYNC_COPY
            async_copy_b128(gk,     &Ks[krow * KP + kcol]);
            async_copy_b128(gk + 8, &Ks[krow * KP + kcol + 8]);
#else
            *(v8bf*)&Ks[krow * KP + kcol]     = *(const v8bf*)(gk);
            *(v8bf*)&Ks[krow * KP + kcol + 8] = *(const v8bf*)(gk + 8);
#endif
            const bf16_t* gv = V + head + (size_t)(key0 + krow) * rs + kcol;
            v8bf v0 = *(const v8bf*)(gv), v1 = *(const v8bf*)(gv + 8);
            #pragma unroll
            for (int e = 0; e < 8; e++) {
                Vt[(kcol + e) * VP + krow]     = v0[e];
                Vt[(kcol + 8 + e) * VP + krow] = v1[e];
            }
#ifdef HAVE_ASYNC_COPY
            WAIT_ASYNC(0);
#endif
        }
        __syncthreads();

        // scores: 16 q-rows x 32 keys
        v8f sacc[2] = {};
        #pragma unroll
        for (int c = 0; c < 4; c++) {
            #pragma unroll
            for (int j = 0; j < 2; j++) {
                V16 kf;
                int n = j * 16 + l16;
                int k0 = c * 32 + lhalf * 8;
                kf.h[0] = *(const v8bf*)&Ks[n * KP + k0];
                kf.h[1] = *(const v8bf*)&Ks[n * KP + k0 + 16];
                sacc[j] = wmma_bf16(qf[c], kf, sacc[j]);
            }
        }

        // online softmax per row (rows 0..7 live in lanes 0-15, 8..15 in 16-31)
        #pragma unroll
        for (int r = 0; r < 8; r++) {
            int m = q0 + r + lhalf * 8;
            float s0 = sacc[0][r] * ATTN_SCALE_;
            float s1 = sacc[1][r] * ATTN_SCALE_;
            if (key0 + l16 > m)      s0 = -3.0e38f;
            if (key0 + 16 + l16 > m) s1 = -3.0e38f;
            float mx = fmaxf(s0, s1);
            #pragma unroll
            for (int off = 1; off < 16; off <<= 1) mx = fmaxf(mx, __shfl_xor(mx, off, 32));
            float mnew = fmaxf(mrow[r], mx);
            float p0 = __expf(s0 - mnew), p1 = __expf(s1 - mnew);
            float ps = p0 + p1;
            #pragma unroll
            for (int off = 1; off < 16; off <<= 1) ps += __shfl_xor(ps, off, 32);
            float alpha = __expf(mrow[r] - mnew);
            lrow[r] = lrow[r] * alpha + ps;
            mrow[r] = mnew;
            #pragma unroll
            for (int d = 0; d < 8; d++) oacc[d][r] *= alpha;
            Ps[wave * (16 * VP) + (r + lhalf * 8) * VP + l16]      = to_bf16(p0);
            Ps[wave * (16 * VP) + (r + lhalf * 8) * VP + 16 + l16] = to_bf16(p1);
        }
        __syncthreads();

        // P as A-fragment, then P*V into O
        V16 pf;
        {
            int k0 = lhalf * 8;
            const bf16_t* pr = &Ps[wave * (16 * VP) + l16 * VP];
            pf.h[0] = *(const v8bf*)(pr + k0);
            pf.h[1] = *(const v8bf*)(pr + k0 + 16);
        }
        #pragma unroll
        for (int d = 0; d < 8; d++) {
            V16 vf;
            int n = d * 16 + l16;
            int k0 = lhalf * 8;
            vf.h[0] = *(const v8bf*)&Vt[n * VP + k0];
            vf.h[1] = *(const v8bf*)&Vt[n * VP + k0 + 16];
            oacc[d] = wmma_bf16(pf, vf, oacc[d]);
        }
    }

    // normalize and write O (bf16, [B,S,H,HD])
    #pragma unroll
    for (int r = 0; r < 8; r++) {
        int m = q0 + r + lhalf * 8;
        float invl = 1.0f / lrow[r];
        bf16_t* orow = O + head + (size_t)m * rs;
        #pragma unroll
        for (int d = 0; d < 8; d++)
            orow[d * 16 + l16] = to_bf16(oacc[d][r] * invl);
    }
}

// ---------------------------------------------------------------------------
// Host-side launch
// ---------------------------------------------------------------------------
extern "C" void kernel_launch(void* const* d_in, const int* in_sizes, int n_in,
                              void* d_out, int out_size, void* d_ws, size_t ws_size,
                              hipStream_t stream) {
    const float* x      = (const float*)d_in[0];
    const float* Wq     = (const float*)d_in[1];
    const float* Wk     = (const float*)d_in[2];
    const float* Wv     = (const float*)d_in[3];
    const float* Wo     = (const float*)d_in[4];
    const float* Wgate  = (const float*)d_in[5];
    const float* Wup    = (const float*)d_in[6];
    const float* Wdown  = (const float*)d_in[7];
    const float* scale1 = (const float*)d_in[8];
    const float* scale2 = (const float*)d_in[9];
    float* out = (float*)d_out;
    char* ws = (char*)d_ws;

    const size_t MB = 1ull << 20;
    bf16_t* WqB    = (bf16_t*)(ws + 0 * MB);     // D*D bf16 = 8 MiB each
    bf16_t* WkB    = (bf16_t*)(ws + 8 * MB);
    bf16_t* WvB    = (bf16_t*)(ws + 16 * MB);
    bf16_t* WoB    = (bf16_t*)(ws + 24 * MB);
    bf16_t* WgateB = (bf16_t*)(ws + 32 * MB);    // F*D bf16 = 32 MiB each
    bf16_t* WupB   = (bf16_t*)(ws + 64 * MB);
    bf16_t* WdownB = (bf16_t*)(ws + 96 * MB);
    bf16_t* hB     = (bf16_t*)(ws + 128 * MB);   // M*D bf16 = 16 MiB each
    bf16_t* qB     = (bf16_t*)(ws + 144 * MB);
    bf16_t* kB     = (bf16_t*)(ws + 160 * MB);
    bf16_t* vB     = (bf16_t*)(ws + 176 * MB);
    bf16_t* aoB    = (bf16_t*)(ws + 192 * MB);
    bf16_t* h2B    = (bf16_t*)(ws + 208 * MB);
    bf16_t* gateB  = (bf16_t*)(ws + 224 * MB);   // M*F bf16 = 64 MiB each
    bf16_t* ffnB   = (bf16_t*)(ws + 288 * MB);   // total 352 MiB

    // 1) weights -> bf16
    k_cast_bf16<<<(D_ * D_) / 1024, 256, 0, stream>>>(Wq, WqB);
    k_cast_bf16<<<(D_ * D_) / 1024, 256, 0, stream>>>(Wk, WkB);
    k_cast_bf16<<<(D_ * D_) / 1024, 256, 0, stream>>>(Wv, WvB);
    k_cast_bf16<<<(D_ * D_) / 1024, 256, 0, stream>>>(Wo, WoB);
    k_cast_bf16<<<(F_ * D_) / 1024, 256, 0, stream>>>(Wgate, WgateB);
    k_cast_bf16<<<(F_ * D_) / 1024, 256, 0, stream>>>(Wup, WupB);
    k_cast_bf16<<<(F_ * D_) / 1024, 256, 0, stream>>>(Wdown, WdownB);

    // 2) h = rmsnorm(x, scale1) -> bf16
    k_rmsnorm_bf16<<<M_, 256, 0, stream>>>(x, scale1, hB);

    // 3) q/k/v = h @ W^T  (bf16 out)
    dim3 gQ(D_ / BN, M_ / BM);
    k_gemm_bf16<1><<<gQ, 256, 0, stream>>>(hB, WqB, qB, nullptr, nullptr, M_, D_, D_);
    k_gemm_bf16<1><<<gQ, 256, 0, stream>>>(hB, WkB, kB, nullptr, nullptr, M_, D_, D_);
    k_gemm_bf16<1><<<gQ, 256, 0, stream>>>(hB, WvB, vB, nullptr, nullptr, M_, D_, D_);

    // 4) RoPE on q, k (in place)
    k_rope_bf16<<<(B_ * S_ * D_ / 2) / 256, 256, 0, stream>>>(qB);
    k_rope_bf16<<<(B_ * S_ * D_ / 2) / 256, 256, 0, stream>>>(kB);

    // 5) flash attention -> aoB
    k_attn<<<dim3(S_ / 128, B_ * H_), 256, 0, stream>>>(qB, kB, vB, aoB);

    // 6) x1 = x + RES_SCALE * (attn @ Wo^T)  -> d_out (fp32)
    k_gemm_bf16<2><<<gQ, 256, 0, stream>>>(aoB, WoB, out, x, nullptr, M_, D_, D_);

    // 7) h2 = rmsnorm(x1, scale2) -> bf16
    k_rmsnorm_bf16<<<M_, 256, 0, stream>>>(out, scale2, h2B);

    // 8) gate = silu(h2 @ Wgate^T) -> bf16
    dim3 gF(F_ / BN, M_ / BM);
    k_gemm_bf16<3><<<gF, 256, 0, stream>>>(h2B, WgateB, gateB, nullptr, nullptr, M_, F_, D_);

    // 9) ffnA = gate * (h2 @ Wup^T) -> bf16
    k_gemm_bf16<4><<<gF, 256, 0, stream>>>(h2B, WupB, ffnB, nullptr, gateB, M_, F_, D_);

    // 10) out = x1 + RES_SCALE * (ffnA @ Wdown^T)   (reads+writes d_out)
    k_gemm_bf16<2><<<gQ, 256, 0, stream>>>(ffnB, WdownB, out, out, nullptr, M_, D_, F_);
}
